// RNN_24635932410038
// MI455X (gfx1250) — compile-verified, matrix operand order checked by reading, not run on hardware
//
#include <hip/hip_runtime.h>
#include <hip/hip_bf16.h>

// ---------------------------------------------------------------------------
// 2-layer SimpleRNN, fused for MI455X (gfx1250, wave32, WMMA bf16 16x16x32).
//   Stage 1: embW = emb @ W0 + b0           (tiny GEMM, result L2-resident)
//   Stage 2: weights U0,W1,U1 -> bf16, transposed [N][K] for B-fragment loads
//   Stage 3: fused recurrence; 1 wave owns 16 batch rows for all 200 steps.
//            Weights live in LDS (96KB of 320KB/WGP); h0/h1 staged in
//            wave-private LDS in bf16; accumulate fp32 via v_wmma.
//            Activations: raw v_exp_f32 + v_rcp_f32, no clamp, no IEEE div.
// ---------------------------------------------------------------------------

typedef __attribute__((ext_vector_type(16))) __bf16 v16bf;
typedef __attribute__((ext_vector_type(8)))  float  v8f;

#define UDIM 128
#define TSEQ 200
#define NW   2            // waves per block
#define ROWS_PER_WAVE 16

union AccU  { v8f   v; float  f[8]; };
union FragU { v16bf v; float4 q[2]; };

#define TWO_LOG2E 2.8853900817779268f   // 2*log2(e)
#define LOG2E     1.4426950408889634f

// tanh(x) = 1 - 2/(2^(x*2*log2e)+1).
// Saturation is handled by float semantics: exp->inf => 1, exp->0 => -1.
__device__ inline float fast_tanh(float x) {
  float e = __builtin_amdgcn_exp2f(x * TWO_LOG2E);   // v_mul + v_exp_f32
  return fmaf(-2.f, __builtin_amdgcn_rcpf(e + 1.f), 1.f);
}

__device__ inline float fast_sigmoid(float x) {
  float e = __builtin_amdgcn_exp2f(-x * LOG2E);
  return __builtin_amdgcn_rcpf(1.f + e);
}

// ---- Stage 1: embW[v][u] = sum_e emb[v][e]*W0[e][u] + b0[u] ----------------
extern "C" __global__ void embw_proj(const float* __restrict__ emb,
                                     const float* __restrict__ W0,
                                     const float* __restrict__ b0,
                                     float* __restrict__ embW) {
  int v = blockIdx.x;
  int u = threadIdx.x;          // 128 threads
  float s = b0[u];
  const float* er = emb + v * UDIM;
#pragma unroll 4
  for (int e = 0; e < UDIM; ++e) s = fmaf(er[e], W0[e * UDIM + u], s);
  embW[v * UDIM + u] = s;
}

// ---- Stage 2: wt[w][n][k] = (bf16) src_w[k][n]  (transpose + convert) ------
extern "C" __global__ void wt_pack(const float* __restrict__ U0,
                                   const float* __restrict__ W1,
                                   const float* __restrict__ U1,
                                   __bf16* __restrict__ wt) {
  int id = blockIdx.x * blockDim.x + threadIdx.x;   // 3*128*128 total
  int w = id >> 14;
  int r = id & 16383;
  int n = r >> 7;
  int k = r & 127;
  const float* src = (w == 0) ? U0 : (w == 1) ? W1 : U1;
  wt[id] = (__bf16)src[k * UDIM + n];
}

// ---- Stage 3: fused recurrence --------------------------------------------
extern "C" __global__ __launch_bounds__(32 * NW) void rnn2_fused(
    const int*   __restrict__ inputs,   // [B][T]
    const float* __restrict__ embW,     // [V][128] fp32
    const __bf16* __restrict__ wtAll,   // [3][128][128] bf16, N-major
    const float* __restrict__ b1,       // [128]
    const float* __restrict__ Wout,     // [128]
    const float* __restrict__ bout,     // [1]
    float*       __restrict__ out) {    // [B]
  extern __shared__ char smem[];
  __bf16* wt  = (__bf16*)smem;                          // 98304 B
  float*  b1s = (float*)(smem + 3 * UDIM * UDIM * 2);   // 512 B
  char* wavemem = smem + 3 * UDIM * UDIM * 2 + 512 +
                  (threadIdx.x >> 5) * (8192 + 4096 + 4096);
  float*  xst  = (float*)wavemem;                 // 16x128 fp32 x_t stage
  __bf16* h0st = (__bf16*)(wavemem + 8192);       // 16x128 bf16
  __bf16* h1st = (__bf16*)(wavemem + 8192 + 4096);

  const int tid  = threadIdx.x;
  const int lane = tid & 31;
  const int l    = lane & 15;   // row (A) / column (B, C/D) index within tile
  const int hi   = lane >> 4;   // which half of the wave
  const int fo   = hi * 8;      // element offset of q[0] within a 32-K chunk
  const int rowbase = (blockIdx.x * NW + (tid >> 5)) * ROWS_PER_WAVE;

  // Cooperative weight load: global -> LDS (b128 copies)
  {
    const uint4* src = (const uint4*)wtAll;
    uint4* dst = (uint4*)wt;
    const int n16 = 3 * UDIM * UDIM * 2 / 16;
    for (int i = tid; i < n16; i += blockDim.x) dst[i] = src[i];
    for (int i = tid; i < UDIM; i += blockDim.x) b1s[i] = b1[i];
  }
  // Zero wave-private h state
  {
    uint4 z = make_uint4(0u, 0u, 0u, 0u);
    uint4* p0 = (uint4*)h0st;
    uint4* p1 = (uint4*)h1st;
    for (int i = lane; i < ROWS_PER_WAVE * UDIM * 2 / 16; i += 32) {
      p0[i] = z; p1[i] = z;
    }
  }
  __syncthreads();

  const __bf16* wtU0 = wt;
  const __bf16* wtW1 = wt + UDIM * UDIM;
  const __bf16* wtU1 = wt + 2 * UDIM * UDIM;

  for (int t = 0; t < TSEQ; ++t) {
    // -- gather this step's input projection rows: embW[token] -> xst (fp32)
    {
      int tok = inputs[(rowbase + l) * TSEQ + t];
      const float4* s = (const float4*)(embW + (size_t)tok * UDIM + hi * 64);
      float4* d = (float4*)(xst + l * UDIM + hi * 64);
      d[0] = s[0]; d[1] = s[1]; d[2] = s[2]; d[3] = s[3];
      if (t + 1 < TSEQ) {   // warm L2/L0 for the next step's rows
        int tok2 = inputs[(rowbase + l) * TSEQ + t + 1];
        __builtin_prefetch(embW + (size_t)tok2 * UDIM + hi * 64, 0, 0);
      }
    }

    // -- layer 0: h0 = tanh(x_t + h0 @ U0) ---------------------------------
    FragU a0[4];
#pragma unroll
    for (int c = 0; c < 4; ++c) {   // A-fragments of old h0 (16x32 per chunk)
      const __bf16* p = h0st + l * UDIM + c * 32 + fo;
      a0[c].q[0] = *(const float4*)p;
      a0[c].q[1] = *(const float4*)(p + 16);
    }
    for (int j = 0; j < 8; ++j) {   // 8 N-tiles of 16 columns
      AccU acc;
#pragma unroll
      for (int r = 0; r < 8; ++r)   // C init = x_t in C/D layout
        acc.f[r] = xst[(r + 8 * hi) * UDIM + j * 16 + l];
      FragU b[4];                   // preload all 4 B chunks, then WMMA chain
#pragma unroll
      for (int c = 0; c < 4; ++c) {
        const __bf16* p = wtU0 + (j * 16 + l) * UDIM + c * 32 + fo;
        b[c].q[0] = *(const float4*)p;
        b[c].q[1] = *(const float4*)(p + 16);
      }
#pragma unroll
      for (int c = 0; c < 4; ++c)
        acc.v = __builtin_amdgcn_wmma_f32_16x16x32_bf16(
            false, a0[c].v, false, b[c].v, (short)0, acc.v, false, false);
#pragma unroll
      for (int r = 0; r < 8; ++r)
        h0st[(r + 8 * hi) * UDIM + j * 16 + l] = (__bf16)fast_tanh(acc.f[r]);
    }

    // -- layer 1: h1 = tanh(h0_new @ W1 + h1 @ U1 + b1) --------------------
    FragU a1[4], a2[4];
#pragma unroll
    for (int c = 0; c < 4; ++c) {
      const __bf16* p0 = h0st + l * UDIM + c * 32 + fo;
      a1[c].q[0] = *(const float4*)p0;
      a1[c].q[1] = *(const float4*)(p0 + 16);
      const __bf16* p1 = h1st + l * UDIM + c * 32 + fo;
      a2[c].q[0] = *(const float4*)p1;
      a2[c].q[1] = *(const float4*)(p1 + 16);
    }
    for (int j = 0; j < 8; ++j) {
      AccU acc;
      float bb = b1s[j * 16 + l];
#pragma unroll
      for (int r = 0; r < 8; ++r) acc.f[r] = bb;
      FragU b[4], b2[4];
#pragma unroll
      for (int c = 0; c < 4; ++c) {
        const __bf16* p = wtW1 + (j * 16 + l) * UDIM + c * 32 + fo;
        b[c].q[0] = *(const float4*)p;
        b[c].q[1] = *(const float4*)(p + 16);
        const __bf16* p2 = wtU1 + (j * 16 + l) * UDIM + c * 32 + fo;
        b2[c].q[0] = *(const float4*)p2;
        b2[c].q[1] = *(const float4*)(p2 + 16);
      }
#pragma unroll
      for (int c = 0; c < 4; ++c)
        acc.v = __builtin_amdgcn_wmma_f32_16x16x32_bf16(
            false, a1[c].v, false, b[c].v, (short)0, acc.v, false, false);
#pragma unroll
      for (int c = 0; c < 4; ++c)
        acc.v = __builtin_amdgcn_wmma_f32_16x16x32_bf16(
            false, a2[c].v, false, b2[c].v, (short)0, acc.v, false, false);
#pragma unroll
      for (int r = 0; r < 8; ++r)
        h1st[(r + 8 * hi) * UDIM + j * 16 + l] = (__bf16)fast_tanh(acc.f[r]);
    }
  }

  // -- final Dense(1) + sigmoid: out[b] = sigmoid(h1[b] . Wout + bout) -----
  {
    float partial = 0.f;
    const __bf16* row = h1st + l * UDIM + hi * 64;
    const float*  w   = Wout + hi * 64;
#pragma unroll 8
    for (int k = 0; k < 64; ++k) partial = fmaf((float)row[k], w[k], partial);
    partial += __shfl_xor(partial, 16, 32);   // combine the two K-halves
    if (hi == 0)
      out[rowbase + l] = fast_sigmoid(partial + bout[0]);
  }
}

// ---------------------------------------------------------------------------
extern "C" void kernel_launch(void* const* d_in, const int* in_sizes, int n_in,
                              void* d_out, int out_size, void* d_ws, size_t ws_size,
                              hipStream_t stream) {
  const int*   inputs = (const int*)  d_in[0];
  const float* emb    = (const float*)d_in[1];
  const float* W0     = (const float*)d_in[2];
  const float* U0     = (const float*)d_in[3];
  const float* b0     = (const float*)d_in[4];
  const float* W1     = (const float*)d_in[5];
  const float* U1     = (const float*)d_in[6];
  const float* b1     = (const float*)d_in[7];
  const float* Wout   = (const float*)d_in[8];
  const float* bout   = (const float*)d_in[9];
  float* out = (float*)d_out;

  const int V = in_sizes[1] / UDIM;          // 10000
  const int B = in_sizes[0] / TSEQ;          // 4096

  // workspace carve: [embW fp32 V*128][wt bf16 3*128*128]
  float*  embW = (float*)d_ws;
  __bf16* wt   = (__bf16*)((char*)d_ws + (size_t)V * UDIM * sizeof(float));

  embw_proj<<<V, UDIM, 0, stream>>>(emb, W0, b0, embW);
  wt_pack<<<(3 * UDIM * UDIM) / 256, 256, 0, stream>>>(U0, W1, U1, wt);

  size_t shmem = 3 * UDIM * UDIM * 2 + 512 + (size_t)NW * (8192 + 4096 + 4096);
  (void)hipFuncSetAttribute((const void*)rnn2_fused,
                            hipFuncAttributeMaxDynamicSharedMemorySize,
                            (int)shmem);
  rnn2_fused<<<B / (ROWS_PER_WAVE * NW), 32 * NW, shmem, stream>>>(
      inputs, embW, wt, b1, Wout, bout, out);

  (void)n_in; (void)out_size; (void)ws_size;
}
